// ViewEncoder_88476326297971
// MI455X (gfx1250) — compile-verified, matrix operand order checked by reading, not run on hardware
//
#include <hip/hip_runtime.h>
#include <hip/hip_bf16.h>

typedef __attribute__((ext_vector_type(16))) __bf16        v16bf;
typedef __attribute__((ext_vector_type(8)))  float         v8f;
typedef __attribute__((ext_vector_type(8)))  unsigned int  v8u;

#define IN_DIM 512
#define HID    64

// round-to-nearest-even f32->bf16 packed pair (host-side layout helper kernels)
__device__ __forceinline__ unsigned int pk_bf16(float x, float y) {
    unsigned int ux = __float_as_uint(x);
    unsigned int uy = __float_as_uint(y);
    unsigned int rx = (ux + 0x7FFFu + ((ux >> 16) & 1u)) >> 16;
    unsigned int ry = (uy + 0x7FFFu + ((uy >> 16) & 1u)) >> 16;
    return (rx & 0xFFFFu) | (ry << 16);
}

__device__ __forceinline__ unsigned short cvt_bf16_u16(float x) {
    unsigned int u = __float_as_uint(x);
    return (unsigned short)((u + 0x7FFFu + ((u >> 16) & 1u)) >> 16);
}

// K-index mapping for 16-bit A/B fragments (ISA 7.12.2, 16-bit A 16x32):
// VGPR j, pair base: j<4 -> 2j ; j>=4 -> 16+2(j-4); lanes 16-31 add 8.
__device__ __forceinline__ int frag_k(int j) {
    return (j < 4) ? (2 * j) : (16 + 2 * (j - 4));
}

// ---------------------------------------------------------------- utilities
__global__ void zero_f32_kernel(float* __restrict__ p, int n) {
    int i = blockIdx.x * blockDim.x + threadIdx.x;
    if (i < n) p[i] = 0.0f;
}

// --------------------------------------------------------------- degree/norm
__global__ void deg_edges_kernel(const long long* __restrict__ dst,
                                 const float* __restrict__ ew,
                                 float* __restrict__ deg, int E) {
    int e = blockIdx.x * blockDim.x + threadIdx.x;
    if (e < E) atomicAdd(&deg[(int)dst[e]], ew[e]);
}

__global__ void deg_finish_kernel(float* __restrict__ deg_dinv, int N) {
    int n = blockIdx.x * blockDim.x + threadIdx.x;
    if (n < N) {
        float d = deg_dinv[n] + 1.0f;          // + self-loop weight 1
        deg_dinv[n] = (d > 0.0f) ? rsqrtf(d) : 0.0f;
    }
}

// ----------------------------------------------------- pack W into B-fragment
// layout: Wpk[((kb*4 + ct)*32 + lane)*8 + j] = pack(W[kk][col], W[kk+1][col])
// col = ct*16 + (lane&15); kk = kb*32 + (lane>>4)*8 + frag_k(j)
__global__ void pack_w_kernel(const float* __restrict__ W,
                              unsigned int* __restrict__ Wpk, int K) {
    int idx = blockIdx.x * blockDim.x + threadIdx.x;
    if (idx >= K * 32) return;
    int j    = idx & 7;
    int lane = (idx >> 3) & 31;
    int ct   = (idx >> 8) & 3;
    int kb   = idx >> 10;
    int col  = ct * 16 + (lane & 15);
    int kk   = kb * 32 + (lane >> 4) * 8 + frag_k(j);
    Wpk[idx] = pk_bf16(W[kk * HID + col], W[(kk + 1) * HID + col]);
}

// ------------------------------------------------- GEMM1: H = X(f32) @ W1
// block = 64 thr = 2 waves. Each wave computes a 16x64 stripe: one shared A
// fragment feeds 4 back-to-back WMMAs (independent accumulators, no RAW).
__global__ __launch_bounds__(64) void gemm_x_kernel(
        const float* __restrict__ X, const unsigned int* __restrict__ Wpk,
        float* __restrict__ H) {
    const int lane = threadIdx.x & 31;
    const int w    = threadIdx.x >> 5;          // 0..1
    const int rowTile = blockIdx.x * 32 + w * 16;
    const int half    = lane >> 4;
    const int l15     = lane & 15;
    const int row     = rowTile + l15;
    const float* xrow = X + (long long)row * IN_DIM;

    v8f c0 = {}, c1 = {}, c2 = {}, c3 = {};
#pragma unroll 1
    for (int kb = 0; kb < IN_DIM / 32; ++kb) {
        const int k0 = kb * 32 + half * 8;
        if (kb + 1 < IN_DIM / 32)
            __builtin_prefetch(xrow + (kb + 1) * 32 + half * 8, 0, 0);
        v16bf a;
#pragma unroll
        for (int j = 0; j < 8; ++j) {
            int kk = k0 + frag_k(j);
            float2 xv = *(const float2*)(xrow + kk);
            a[2 * j]     = (__bf16)xv.x;        // -> v_cvt_pk_bf16_f32
            a[2 * j + 1] = (__bf16)xv.y;
        }
        const unsigned int* wb = Wpk + ((kb * 4 * 32 + lane) << 3);
        v16bf b0 = __builtin_bit_cast(v16bf, *(const v8u*)(wb + 0 * 256));
        v16bf b1 = __builtin_bit_cast(v16bf, *(const v8u*)(wb + 1 * 256));
        v16bf b2 = __builtin_bit_cast(v16bf, *(const v8u*)(wb + 2 * 256));
        v16bf b3 = __builtin_bit_cast(v16bf, *(const v8u*)(wb + 3 * 256));
        c0 = __builtin_amdgcn_wmma_f32_16x16x32_bf16(false, a, false, b0,
                                                     (short)0, c0, false, false);
        c1 = __builtin_amdgcn_wmma_f32_16x16x32_bf16(false, a, false, b1,
                                                     (short)0, c1, false, false);
        c2 = __builtin_amdgcn_wmma_f32_16x16x32_bf16(false, a, false, b2,
                                                     (short)0, c2, false, false);
        c3 = __builtin_amdgcn_wmma_f32_16x16x32_bf16(false, a, false, b3,
                                                     (short)0, c3, false, false);
    }
    float* hb = H + (long long)(rowTile + half * 8) * HID + l15;
#pragma unroll
    for (int r = 0; r < 8; ++r) {
        hb[(long long)r * HID + 0]  = c0[r];
        hb[(long long)r * HID + 16] = c1[r];
        hb[(long long)r * HID + 32] = c2[r];
        hb[(long long)r * HID + 48] = c3[r];
    }
}

// --------------------------------------- GEMM2: H2 = Hbf16(row-major) @ W2
__global__ __launch_bounds__(64) void gemm_h_kernel(
        const unsigned int* __restrict__ Apk,  // bf16 pairs, row-major [N][32]
        const unsigned int* __restrict__ Wpk,
        float* __restrict__ H2) {
    const int lane = threadIdx.x & 31;
    const int w    = threadIdx.x >> 5;
    const int rowTile = blockIdx.x * 32 + w * 16;
    const int half    = lane >> 4;
    const int l15     = lane & 15;
    const int row     = rowTile + l15;
    const unsigned int* arow = Apk + (long long)row * (HID / 2);

    v8f c0 = {}, c1 = {}, c2 = {}, c3 = {};
#pragma unroll
    for (int kb = 0; kb < HID / 32; ++kb) {
        const int k0 = kb * 32 + half * 8;
        v8u au;
#pragma unroll
        for (int j = 0; j < 8; ++j) {
            int kk = k0 + frag_k(j);
            au[j] = arow[kk >> 1];       // pair (kk, kk+1) is one dword
        }
        v16bf a = __builtin_bit_cast(v16bf, au);
        const unsigned int* wb = Wpk + ((kb * 4 * 32 + lane) << 3);
        v16bf b0 = __builtin_bit_cast(v16bf, *(const v8u*)(wb + 0 * 256));
        v16bf b1 = __builtin_bit_cast(v16bf, *(const v8u*)(wb + 1 * 256));
        v16bf b2 = __builtin_bit_cast(v16bf, *(const v8u*)(wb + 2 * 256));
        v16bf b3 = __builtin_bit_cast(v16bf, *(const v8u*)(wb + 3 * 256));
        c0 = __builtin_amdgcn_wmma_f32_16x16x32_bf16(false, a, false, b0,
                                                     (short)0, c0, false, false);
        c1 = __builtin_amdgcn_wmma_f32_16x16x32_bf16(false, a, false, b1,
                                                     (short)0, c1, false, false);
        c2 = __builtin_amdgcn_wmma_f32_16x16x32_bf16(false, a, false, b2,
                                                     (short)0, c2, false, false);
        c3 = __builtin_amdgcn_wmma_f32_16x16x32_bf16(false, a, false, b3,
                                                     (short)0, c3, false, false);
    }
    float* hb = H2 + (long long)(rowTile + half * 8) * HID + l15;
#pragma unroll
    for (int r = 0; r < 8; ++r) {
        hb[(long long)r * HID + 0]  = c0[r];
        hb[(long long)r * HID + 16] = c1[r];
        hb[(long long)r * HID + 32] = c2[r];
        hb[(long long)r * HID + 48] = c3[r];
    }
}

// ----------------------------------- edge scatter: acc[dst] += norm * H[src]
// one thread = (edge, 16-channel group); 4 float4 gathers + 16 f32 atomics
__global__ void scatter_kernel(const long long* __restrict__ src,
                               const long long* __restrict__ dst,
                               const float* __restrict__ ew,
                               const float* __restrict__ dinv,
                               const float* __restrict__ Hin,
                               float* __restrict__ acc, int E) {
    int tid = blockIdx.x * blockDim.x + threadIdx.x;
    if (tid >= E * 4) return;
    int e = tid >> 2;
    int g = (tid & 3) * 16;
    int s = (int)src[e];
    int d = (int)dst[e];
    float norm = dinv[s] * ew[e] * dinv[d];
    const float4* hp = (const float4*)(Hin + (long long)s * HID + g);
    float* ap = acc + (long long)d * HID + g;
#pragma unroll
    for (int q = 0; q < 4; ++q) {
        float4 v = hp[q];
        atomicAdd(ap + 4 * q + 0, v.x * norm);
        atomicAdd(ap + 4 * q + 1, v.y * norm);
        atomicAdd(ap + 4 * q + 2, v.z * norm);
        atomicAdd(ap + 4 * q + 3, v.w * norm);
    }
}

// ---------------- finish layer1: add self-loop + bias, ReLU, emit f32 + bf16
__global__ void finish1_kernel(const float* __restrict__ acc,
                               const float* __restrict__ H,
                               const float* __restrict__ dinv,
                               const float* __restrict__ bias,
                               float* __restrict__ Hf,
                               unsigned short* __restrict__ Hb, int total) {
    int tid = blockIdx.x * blockDim.x + threadIdx.x;
    if (tid >= total) return;
    int c = tid & (HID - 1);
    int n = tid >> 6;
    float di = dinv[n];
    float v = acc[tid] + H[tid] * di * di + bias[c];
    v = fmaxf(v, 0.0f);
    Hf[tid] = v;
    Hb[tid] = cvt_bf16_u16(v);
}

// ---------------- finish layer2: add self-loop + bias (no ReLU) -> output
__global__ void finish2_kernel(const float* __restrict__ acc,
                               const float* __restrict__ H,
                               const float* __restrict__ dinv,
                               const float* __restrict__ bias,
                               float* __restrict__ out, int total) {
    int tid = blockIdx.x * blockDim.x + threadIdx.x;
    if (tid >= total) return;
    int c = tid & (HID - 1);
    int n = tid >> 6;
    float di = dinv[n];
    out[tid] = acc[tid] + H[tid] * di * di + bias[c];
}

extern "C" void kernel_launch(void* const* d_in, const int* in_sizes, int n_in,
                              void* d_out, int out_size, void* d_ws, size_t ws_size,
                              hipStream_t stream) {
    const float*     x   = (const float*)d_in[0];
    const long long* ei  = (const long long*)d_in[1];   // int64 edge_index [2,E]
    const float*     ew  = (const float*)d_in[2];
    const float*     W1  = (const float*)d_in[3];
    const float*     b1  = (const float*)d_in[4];
    const float*     W2  = (const float*)d_in[5];
    const float*     b2  = (const float*)d_in[6];
    float*           out = (float*)d_out;

    const int N  = in_sizes[0] / IN_DIM;    // 100000
    const int E  = in_sizes[1] / 2;         // 1600000
    const int ND = N * HID;                 // 6.4M

    const long long* src = ei;
    const long long* dst = ei + E;

    // ---- workspace layout (256B aligned slabs) ----
    char* wp = (char*)d_ws;
    auto carve = [&](size_t bytes) {
        char* p = wp;
        wp += (bytes + 255) & ~(size_t)255;
        return p;
    };
    float*          dinv  = (float*)carve((size_t)N * 4);
    unsigned int*   w1pk  = (unsigned int*)carve((size_t)IN_DIM * 32 * 4);
    unsigned int*   w2pk  = (unsigned int*)carve((size_t)HID * 32 * 4);
    float*          h1    = (float*)carve((size_t)ND * 4);
    float*          acc   = (float*)carve((size_t)ND * 4);
    float*          h1p   = (float*)carve((size_t)ND * 4);
    unsigned short* h1pb  = (unsigned short*)carve((size_t)ND * 2);
    float*          h2    = (float*)carve((size_t)ND * 4);
    (void)ws_size;

    const int T = 256;
    auto cdiv = [](int a, int b) { return (a + b - 1) / b; };

    // degrees -> dinv
    zero_f32_kernel<<<cdiv(N, T), T, 0, stream>>>(dinv, N);
    deg_edges_kernel<<<cdiv(E, T), T, 0, stream>>>(dst, ew, dinv, E);
    deg_finish_kernel<<<cdiv(N, T), T, 0, stream>>>(dinv, N);

    // pack weights into bf16 B-fragments
    pack_w_kernel<<<cdiv(IN_DIM * 32, T), T, 0, stream>>>(W1, w1pk, IN_DIM);
    pack_w_kernel<<<cdiv(HID * 32, T), T, 0, stream>>>(W2, w2pk, HID);

    // layer 1: transform (WMMA), propagate, finish(ReLU + bias)
    gemm_x_kernel<<<N / 32, 64, 0, stream>>>(x, w1pk, h1);
    zero_f32_kernel<<<cdiv(ND, T), T, 0, stream>>>(acc, ND);
    scatter_kernel<<<cdiv(E * 4, T), T, 0, stream>>>(src, dst, ew, dinv, h1, acc, E);
    finish1_kernel<<<cdiv(ND, T), T, 0, stream>>>(acc, h1, dinv, b1, h1p, h1pb, ND);

    // layer 2: transform (WMMA), propagate, finish(bias)
    gemm_h_kernel<<<N / 32, 64, 0, stream>>>((const unsigned int*)h1pb, w2pk, h2);
    zero_f32_kernel<<<cdiv(ND, T), T, 0, stream>>>(acc, ND);
    scatter_kernel<<<cdiv(E * 4, T), T, 0, stream>>>(src, dst, ew, dinv, h1p, acc, E);
    finish2_kernel<<<cdiv(ND, T), T, 0, stream>>>(acc, h2, dinv, b2, out, ND);
}
